// BatchEncoder_24068996726969
// MI455X (gfx1250) — matrix operand order repaired and checked by reading, not compiled, not run on hardware
//
#include <hip/hip_runtime.h>

// ---------------- CDNA5 / gfx1250 grouped-MLP kernel ----------------
// nodes gather -> bf16 -> [X @ W1 + b1, relu] -> [H @ W2 + b2] -> scatter
// Matrix math on v_wmma_f32_16x16x32_bf16 (wave32 WMMA).
// Next tile's random gather is warmed with global_prefetch_b8 (zero VGPR /
// zero counter cost), with a 2-deep pipeline of gather indices, so the
// blocking gather at the top of each tile hits WGP$/L2 instead of HBM.

#define E      128
#define TWO_E  256
#define BM     64     // M rows per tile
#define LDA    264    // padded LDS leading dim (elements) to avoid bank conflicts

typedef __attribute__((ext_vector_type(16))) __bf16 bf16x16;
typedef __attribute__((ext_vector_type(8)))  __bf16 bf16x8;
typedef __attribute__((ext_vector_type(8)))  float  f32x8;
typedef __attribute__((ext_vector_type(4)))  float  f32x4;

__device__ __forceinline__ unsigned short f2bf(float f) {
  union { float f; unsigned u; } x; x.f = f;
  unsigned r = x.u + 0x7FFFu + ((x.u >> 16) & 1u);   // round-to-nearest-even
  return (unsigned short)(r >> 16);
}
__device__ __forceinline__ unsigned pack2(float a, float b) {
  return (unsigned)f2bf(a) | ((unsigned)f2bf(b) << 16);
}

// A fragment: 16x32 bf16, row-major LDS tile (ld = LDA).
// Lanes 0-15: row M=lane, K {kk+0..7, kk+16..23}; lanes 16-31: K {kk+8..15, kk+24..31}.
__device__ __forceinline__ bf16x16 frag_a(const unsigned short* base, int row0, int kk, int lane) {
  int r  = row0 + (lane & 15);
  int kb = kk + ((lane >> 4) << 3);
  const unsigned short* p = base + r * LDA + kb;
  bf16x8 lo = *(const bf16x8*)(p);
  bf16x8 hi = *(const bf16x8*)(p + 16);
  return __builtin_shufflevector(lo, hi, 0,1,2,3,4,5,6,7,8,9,10,11,12,13,14,15);
}

// B fragment: 32x16 bf16 from column-major-staged weights sW[n][k] (ld = LDA).
// Lane n (0-15): column col0+n, K {kk+0..7, kk+16..23}; lane n+16: same column, K {kk+8..15, kk+24..31}.
__device__ __forceinline__ bf16x16 frag_b(const unsigned short* baseNK, int col0, int kk, int lane) {
  int c  = col0 + (lane & 15);
  int kb = kk + ((lane >> 4) << 3);
  const unsigned short* p = baseNK + c * LDA + kb;
  bf16x8 lo = *(const bf16x8*)(p);
  bf16x8 hi = *(const bf16x8*)(p + 16);
  return __builtin_shufflevector(lo, hi, 0,1,2,3,4,5,6,7,8,9,10,11,12,13,14,15);
}

// -------- kernel 1: out = nodes (streamed, nontemporal) --------
__global__ void copy_f4_kernel(const f32x4* __restrict__ src, f32x4* __restrict__ dst, size_t n4) {
  size_t i = (size_t)blockIdx.x * blockDim.x + threadIdx.x;
  if (i < n4) {
    f32x4 v = __builtin_nontemporal_load(&src[i]);
    __builtin_nontemporal_store(v, &dst[i]);
  }
}

// -------- kernel 2: grouped MLP + scatter --------
__global__ void __launch_bounds__(256, 1)
mlp_wmma_kernel(const float* __restrict__ nodes,
                const float* __restrict__ W1, const float* __restrict__ b1,
                const float* __restrict__ W2, const float* __restrict__ b2,
                const int*   __restrict__ idx, const int* __restrict__ out_idx,
                float* __restrict__ out, int MT)
{
  extern __shared__ unsigned char smem_raw[];
  unsigned short* sW1 = (unsigned short*)smem_raw;      // [256][LDA] bf16, layout [n][k]
  unsigned short* sW2 = sW1 + TWO_E * LDA;              // [128][LDA] bf16, layout [n][k]
  unsigned short* sX  = sW2 + E * LDA;                  // [64][LDA]  bf16, row-major
  unsigned short* sH  = sX  + BM * LDA;                 // [64][LDA]  bf16, row-major
  float* sB1 = (float*)(sH + BM * LDA);                 // [256]
  float* sB2 = sB1 + TWO_E;                             // [128]

  const int tid   = threadIdx.x;
  const int lane  = tid & 31;
  const int wave  = tid >> 5;            // 8 waves: (wave&3)=M strip, (wave>>2)=N strip
  const int t     = blockIdx.y;          // type
  const int mrow0 = (wave & 3) * 16;

  // gather-staging role of this thread (constant): 64 rows x 2 operands x 2 halves
  const int gr    = tid >> 2;
  const int gop   = (tid >> 1) & 1;
  const int gpart = tid & 1;

  // ---- stage this type's weights once per block: fp32 -> bf16, transpose to [n][k] ----
  {
    const float4* w1v = (const float4*)(W1 + (size_t)t * TWO_E * TWO_E);
    for (int id = tid; id < (TWO_E * TWO_E) / 4; id += 256) {
      float4 v = w1v[id];
      int k  = id >> 6;            // row (K) in W1
      int n0 = (id & 63) << 2;     // 4 consecutive columns (N)
      sW1[(n0 + 0) * LDA + k] = f2bf(v.x);
      sW1[(n0 + 1) * LDA + k] = f2bf(v.y);
      sW1[(n0 + 2) * LDA + k] = f2bf(v.z);
      sW1[(n0 + 3) * LDA + k] = f2bf(v.w);
    }
    const float4* w2v = (const float4*)(W2 + (size_t)t * TWO_E * E);
    for (int id = tid; id < (TWO_E * E) / 4; id += 256) {
      float4 v = w2v[id];
      int k  = id >> 5;
      int n0 = (id & 31) << 2;
      sW2[(n0 + 0) * LDA + k] = f2bf(v.x);
      sW2[(n0 + 1) * LDA + k] = f2bf(v.y);
      sW2[(n0 + 2) * LDA + k] = f2bf(v.z);
      sW2[(n0 + 3) * LDA + k] = f2bf(v.w);
    }
    if (tid < TWO_E) sB1[tid] = b1[(size_t)t * TWO_E + tid];
    if (tid < E)     sB2[tid] = b2[(size_t)t * E + tid];
  }

  const int tilesPerType = MT / BM;
  const int tile0  = blockIdx.x;
  const int stride = gridDim.x;
  if (tile0 >= tilesPerType) return;

  // 2-deep pipeline of gather indices (clamped; duplicate loads are harmless)
  const int lastTile = tilesPerType - 1;
  const size_t idxBase = (size_t)t * MT;
  int tB = tile0 + stride; if (tB > lastTile) tB = lastTile;
  int nodeA = idx[(idxBase + (size_t)tile0 * BM + gr) * 2 + gop];  // this tile
  int nodeB = idx[(idxBase + (size_t)tB    * BM + gr) * 2 + gop];  // next tile

  for (int tile = tile0; tile < tilesPerType; tile += stride) {
    const int m0 = tile * BM;

    // ---- stage current X tile (rows warmed by last iteration's prefetch) ----
    {
      const float4* src = (const float4*)(nodes + (size_t)nodeA * E) + gpart * 16;
      unsigned short* dst = sX + gr * LDA + gop * E + gpart * 64;
#pragma unroll
      for (int i = 0; i < 16; ++i) {
        float4 v = src[i];
        uint2 pk; pk.x = pack2(v.x, v.y); pk.y = pack2(v.z, v.w);
        *(uint2*)(dst + i * 4) = pk;
      }
    }

    // ---- issue idx load for tile+2 (consumed next iteration, no wait here) ----
    int tC = tile + 2 * stride; if (tC > lastTile) tC = lastTile;
    int nodeC = idx[(idxBase + (size_t)tC * BM + gr) * 2 + gop];

    // ---- warm next tile's gather: global_prefetch_b8, zero VGPR/counter cost ----
    {
      const float* p = nodes + (size_t)nodeB * E + gpart * 64;
      __builtin_prefetch(p, 0, 1);        // first 128B line of this thread's slice
      __builtin_prefetch(p + 32, 0, 1);   // second 128B line
    }
    __syncthreads();

    // ---- layer 1: H = relu(X @ W1 + b1); wave computes 16 x 128 ----
    {
      const int ncol0 = (wave >> 2) * 128;
      f32x8 acc[8] = {};
      for (int kk = 0; kk < TWO_E; kk += 32) {
        bf16x16 a = frag_a(sX, mrow0, kk, lane);
#pragma unroll
        for (int j = 0; j < 8; ++j) {
          bf16x16 b = frag_b(sW1, ncol0 + j * 16, kk, lane);
          acc[j] = __builtin_amdgcn_wmma_f32_16x16x32_bf16(
              false, a, false, b, (short)0, acc[j], false, false);
        }
      }
      // bias + relu, store bf16 H tile
      const int hi8 = (lane >> 4) << 3;
      const int cb  = ncol0 + (lane & 15);
#pragma unroll
      for (int j = 0; j < 8; ++j) {
        int n = cb + j * 16;
        float bias = sB1[n];
#pragma unroll
        for (int i = 0; i < 8; ++i) {
          float v = fmaxf(acc[j][i] + bias, 0.0f);
          sH[(mrow0 + i + hi8) * LDA + n] = f2bf(v);
        }
      }
    }
    __syncthreads();   // sH ready for layer 2; also fences next iter's sX rewrite

    // ---- layer 2: OUT = H @ W2 + b2; wave computes 16 x 64; scatter ----
    {
      const int ncol0 = (wave >> 2) * 64;
      f32x8 acc[4] = {};
      for (int kk = 0; kk < TWO_E; kk += 32) {
        bf16x16 a = frag_a(sH, mrow0, kk, lane);
#pragma unroll
        for (int j = 0; j < 4; ++j) {
          bf16x16 b = frag_b(sW2, ncol0 + j * 16, kk, lane);
          acc[j] = __builtin_amdgcn_wmma_f32_16x16x32_bf16(
              false, a, false, b, (short)0, acc[j], false, false);
        }
      }
      const int hi8 = (lane >> 4) << 3;
      const int cb  = ncol0 + (lane & 15);
      int grow[8];
#pragma unroll
      for (int i = 0; i < 8; ++i)
        grow[i] = out_idx[(size_t)t * MT + m0 + mrow0 + i + hi8];
#pragma unroll
      for (int j = 0; j < 4; ++j) {
        int n = cb + j * 16;
        float bias = sB2[n];
#pragma unroll
        for (int i = 0; i < 8; ++i)
          out[(size_t)grow[i] * E + n] = acc[j][i] + bias;
      }
    }
    // no end barrier needed: next staging only writes sX (layer-1 readers are
    // past the mid barrier), and next sH writes are fenced by the next
    // pre-layer-1 __syncthreads().
    nodeA = nodeB;
    nodeB = nodeC;
  }
}

extern "C" void kernel_launch(void* const* d_in, const int* in_sizes, int n_in,
                              void* d_out, int out_size, void* d_ws, size_t ws_size,
                              hipStream_t stream) {
  const float* nodes = (const float*)d_in[0];
  const float* W1    = (const float*)d_in[1];
  const float* b1    = (const float*)d_in[2];
  const float* W2    = (const float*)d_in[3];
  const float* b2    = (const float*)d_in[4];
  const int*   idx   = (const int*)d_in[5];
  const int*   oidx  = (const int*)d_in[6];
  float*       out   = (float*)d_out;

  const int T  = in_sizes[1] / (TWO_E * TWO_E);
  const int MT = in_sizes[5] / (2 * T);

  // 1) out = nodes (rows not in out_idx keep their value; scatter overwrites the rest)
  size_t n4 = (size_t)in_sizes[0] / 4;
  unsigned cblocks = (unsigned)((n4 + 255) / 256);
  copy_f4_kernel<<<cblocks, 256, 0, stream>>>((const f32x4*)nodes, (f32x4*)out, n4);

  // 2) grouped MLP, persistent blocks per type
  dim3 grid(128, (unsigned)T);
  size_t shmem = (size_t)(TWO_E + E + 2 * BM) * LDA * sizeof(unsigned short)
               + (size_t)(TWO_E + E) * sizeof(float);   // 271,872 B < 320 KB/WGP
  mlp_wmma_kernel<<<grid, 256, shmem, stream>>>(nodes, W1, b1, W2, b2, idx, oidx, out, MT);
}
